// MultiHeadSelfAttention_79723182948666
// MI455X (gfx1250) — matrix operand order repaired and checked by reading, not compiled, not run on hardware
//
#include <hip/hip_runtime.h>
#include <hip/hip_bf16.h>

typedef __attribute__((ext_vector_type(16))) _Float16 v16h;
typedef __attribute__((ext_vector_type(8)))  _Float16 v8h;
typedef __attribute__((ext_vector_type(8)))  float    v8f;

namespace {
constexpr int Bn  = 4;
constexpr int Sn  = 2048;
constexpr int Dn  = 1024;
constexpr int Hn  = 16;
constexpr int HDn = 64;
constexpr int Mn  = Bn * Sn;                   // 8192 rows for projection GEMMs
constexpr size_t XEL = (size_t)Bn * Sn * Dn;   // 8388608 elements
constexpr size_t WEL = (size_t)Dn * Dn;        // 1048576 elements
}

union V16u { v16h v; v8h h[2]; };

__device__ inline v8f wmma32(v16h a, v16h b, v8f c) {
  return __builtin_amdgcn_wmma_f32_16x16x32_f16(false, a, false, b, (short)0, c,
                                                false, false);
}

// A-operand fragment (16x32 f16) from a per-lane base pointer:
// lane<16 -> row=lane,   halves {0..7, 16..23} relative to p
// lane>=16 -> row=lane-16 via caller base, halves shifted by caller (+8)
__device__ inline v16h ldA(const _Float16* p) {
  V16u u;
  u.h[0] = *(const v8h*)(p);
  u.h[1] = *(const v8h*)(p + 16);
  return u.v;
}

// B-operand fragment (32x16 f16): halves {0..7, 16..23} relative to p (caller shifts +8 for lane>=16).
__device__ inline v16h ldB(const _Float16* p) {
  V16u u;
  u.h[0] = *(const v8h*)(p);
  u.h[1] = *(const v8h*)(p + 16);
  return u.v;
}

template <int MASK>
__device__ inline float swz_xor(float v) {
  int i = __builtin_amdgcn_ds_swizzle(__builtin_bit_cast(int, v),
                                      (MASK << 10) | 0x1F);
  return __builtin_bit_cast(float, i);
}
__device__ inline float rmax16(float v) {
  v = fmaxf(v, swz_xor<1>(v));
  v = fmaxf(v, swz_xor<2>(v));
  v = fmaxf(v, swz_xor<4>(v));
  v = fmaxf(v, swz_xor<8>(v));
  return v;
}
__device__ inline float rsum16(float v) {
  v += swz_xor<1>(v);
  v += swz_xor<2>(v);
  v += swz_xor<4>(v);
  v += swz_xor<8>(v);
  return v;
}

// ---------------------------------------------------------------------------
// f32 -> f16 convert (grid-stride)
__global__ void cvt_f16_kernel(const float* __restrict__ src,
                               _Float16* __restrict__ dst, int n) {
  int i = blockIdx.x * blockDim.x + threadIdx.x;
  int stride = gridDim.x * blockDim.x;
  for (; i < n; i += stride) dst[i] = (_Float16)src[i];
}

// f32 W[k][n] -> f16 Wt[n][k] (transposed, so B fragments are contiguous loads)
__global__ void cvt_wt_kernel(const float* __restrict__ src,
                              _Float16* __restrict__ dst) {
  int i = blockIdx.x * blockDim.x + threadIdx.x;   // i = n*Dn + k
  int n = i >> 10;
  int k = i & (Dn - 1);
  dst[i] = (_Float16)src[(size_t)k * Dn + n];
}

// ---------------------------------------------------------------------------
// GEMM: out[m,n] = A[m,:] @ Wt[n,:]^T + bias[n]
// Wave tile 32x64 (8 accs), block = 8 waves as 4(m) x 2(n) -> 128x128 tile.
// Single A/B base pointer per wave; all fragments via constant IOFFSETs.
// mode 0: f16 out, layout [B,H,S,HD]   (Q, K)
// mode 1: f16 out, layout [B,H,HD,S]   (V transposed)
// mode 2: f32 out, layout [M,N]        (final O projection)
__global__ void __launch_bounds__(256)
gemm_wmma_kernel(const _Float16* __restrict__ A,
                 const _Float16* __restrict__ Wt,
                 const float* __restrict__ bias,
                 void* __restrict__ outp, int mode) {
  const int lane = threadIdx.x & 31;
  const int wave = threadIdx.x >> 5;
  const int wm = wave & 3;
  const int wn = wave >> 2;
  const int m0 = blockIdx.x * 128 + wm * 32;
  const int n0 = blockIdx.y * 128 + wn * 64;

  // per-lane base pointers; every other fragment is a constant offset away
  const _Float16* pA = A  + (size_t)(m0 + (lane & 15)) * Dn + ((lane >> 4) << 3);
  const _Float16* pB = Wt + (size_t)(n0 + (lane & 15)) * Dn + ((lane >> 4) << 3);

  v8f acc[8] = {};

#pragma unroll 4
  for (int k0 = 0; k0 < Dn; k0 += 32) {
    v16h a0 = ldA(pA + k0);
    v16h a1 = ldA(pA + k0 + 16 * Dn);
    v16h b0 = ldB(pB + k0);
    v16h b1 = ldB(pB + k0 + 16 * Dn);
    v16h b2 = ldB(pB + k0 + 32 * Dn);
    v16h b3 = ldB(pB + k0 + 48 * Dn);

    acc[0] = wmma32(a0, b0, acc[0]);
    acc[1] = wmma32(a0, b1, acc[1]);
    acc[2] = wmma32(a0, b2, acc[2]);
    acc[3] = wmma32(a0, b3, acc[3]);
    acc[4] = wmma32(a1, b0, acc[4]);
    acc[5] = wmma32(a1, b1, acc[5]);
    acc[6] = wmma32(a1, b2, acc[6]);
    acc[7] = wmma32(a1, b3, acc[7]);
  }

  const int rhalf = (lane >> 4) << 3;   // C layout: lane>=16 holds rows +8
#pragma unroll
  for (int half = 0; half < 2; ++half) {
#pragma unroll
    for (int nt = 0; nt < 4; ++nt) {
      const int n = n0 + nt * 16 + (lane & 15);
      const float bv = bias[n];
#pragma unroll
      for (int r = 0; r < 8; ++r) {
        const int m = m0 + half * 16 + r + rhalf;
        const float val = acc[half * 4 + nt][r] + bv;
        if (mode == 2) {
          ((float*)outp)[(size_t)m * Dn + n] = val;
        } else {
          const int bb = m >> 11;          // m / Sn
          const int ss = m & (Sn - 1);
          const int hh = n >> 6;           // n / HDn
          const int hd = n & (HDn - 1);
          _Float16* o = (_Float16*)outp;
          if (mode == 0)
            o[(((size_t)(bb * Hn + hh)) * Sn + ss) * HDn + hd] = (_Float16)val;
          else
            o[(((size_t)(bb * Hn + hh)) * HDn + hd) * Sn + ss] = (_Float16)val;
        }
      }
    }
  }
}

// ---------------------------------------------------------------------------
// Flash attention: each wave owns 16 query rows of one (b,h).
// Q,K: [B,H,S,HD] f16.  Vt: [B,H,HD,S] f16.  mask: (1,1,S,S) f32 multiplicative.
// ctx out: [B,S,D] f16.
// All K/V/mask addresses derive from three base pointers + constant offsets.
__global__ void __launch_bounds__(256)
attn_wmma_kernel(const _Float16* __restrict__ Qh,
                 const _Float16* __restrict__ Kh,
                 const _Float16* __restrict__ VTh,
                 const float* __restrict__ mask,
                 _Float16* __restrict__ ctx) {
  __shared__ _Float16 Plds[8][16][32];   // per-wave 16x32 P tile (8 KB)

  const int lane = threadIdx.x & 31;
  const int wave = threadIdx.x >> 5;
  const int bh = blockIdx.x;                       // b*H + h
  const int q0 = blockIdx.y * 128 + wave * 16;
  const int bb = bh >> 4;
  const int hh = bh & 15;
  const int rhalf = (lane >> 4) << 3;

  const _Float16* Qbase = Qh + (size_t)bh * Sn * HDn;
  // per-lane base pointers
  const _Float16* pK = Kh + (size_t)bh * Sn * HDn
                          + (size_t)(lane & 15) * HDn + ((lane >> 4) << 3);
  const _Float16* pV = VTh + (size_t)bh * HDn * Sn
                           + (size_t)(lane & 15) * Sn + ((lane >> 4) << 3);
  const float* pM = mask + (size_t)(q0 + rhalf) * Sn + (lane & 15);

  // Q fragments for the 16x64 tile (two K-chunks of 32), loaded once
  const _Float16* pQ = Qbase + (size_t)(q0 + (lane & 15)) * HDn + ((lane >> 4) << 3);
  const v16h aq0 = ldA(pQ);
  const v16h aq1 = ldA(pQ + 32);

  v8f acc[4] = {};
  float mrow[8], lrow[8];
#pragma unroll
  for (int r = 0; r < 8; ++r) { mrow[r] = -1e30f; lrow[r] = 0.f; }

  for (int j = 0; j < Sn; j += 32) {
    // ---- issue all loads for this block up front (K, V, mask)
    const _Float16* pKj = pK + (size_t)j * HDn;
    v16h k00 = ldB(pKj);
    v16h k01 = ldB(pKj + 32);
    v16h k10 = ldB(pKj + 16 * HDn);
    v16h k11 = ldB(pKj + 16 * HDn + 32);

    v16h vb0 = ldB(pV + j);
    v16h vb1 = ldB(pV + j + 16 * Sn);
    v16h vb2 = ldB(pV + j + 32 * Sn);
    v16h vb3 = ldB(pV + j + 48 * Sn);

    float mval[2][8];
#pragma unroll
    for (int t = 0; t < 2; ++t)
#pragma unroll
      for (int r = 0; r < 8; ++r)
        mval[t][r] = pM[(size_t)r * Sn + j + t * 16];

    // ---- scores 16x32 (two 16x16 WMMA tiles)
    v8f sc[2];
    {
      v8f c = {};
      c = wmma32(aq0, k00, c);
      c = wmma32(aq1, k01, c);
      sc[0] = c;
    }
    {
      v8f c = {};
      c = wmma32(aq0, k10, c);
      c = wmma32(aq1, k11, c);
      sc[1] = c;
    }
#pragma unroll
    for (int t = 0; t < 2; ++t)
#pragma unroll
      for (int r = 0; r < 8; ++r)
        sc[t][r] = sc[t][r] * 0.125f * mval[t][r];   // 1/sqrt(64) then mask

    // ---- online softmax over this 32-wide key block
    float fac[8];
#pragma unroll
    for (int r = 0; r < 8; ++r) {
      const float mx = fmaxf(rmax16(sc[0][r]), rmax16(sc[1][r]));
      const float mnew = fmaxf(mrow[r], mx);
      fac[r] = __expf(mrow[r] - mnew);
      mrow[r] = mnew;
      const float p0 = __expf(sc[0][r] - mnew);
      const float p1 = __expf(sc[1][r] - mnew);
      lrow[r] = lrow[r] * fac[r] + rsum16(p0 + p1);
      // stash P (C-layout) into this wave's private LDS slice as f16
      Plds[wave][r + rhalf][(lane & 15)]      = (_Float16)p0;
      Plds[wave][r + rhalf][16 + (lane & 15)] = (_Float16)p1;
    }
#pragma unroll
    for (int nt = 0; nt < 4; ++nt)
#pragma unroll
      for (int r = 0; r < 8; ++r) acc[nt][r] *= fac[r];

    // ---- P (A fragment, via LDS relayout) @ V -> ctx accumulation
    {
      const int row = lane & 15;
      const int kb  = (lane >> 4) << 3;
      const _Float16* p = &Plds[wave][row][kb];
      V16u u;
      u.h[0] = *(const v8h*)(p);
      u.h[1] = *(const v8h*)(p + 16);
      const v16h pa = u.v;
      acc[0] = wmma32(pa, vb0, acc[0]);
      acc[1] = wmma32(pa, vb1, acc[1]);
      acc[2] = wmma32(pa, vb2, acc[2]);
      acc[3] = wmma32(pa, vb3, acc[3]);
    }
  }

  // ---- normalize and store ctx (f16, [B,S,D])
#pragma unroll
  for (int r = 0; r < 8; ++r) lrow[r] = 1.0f / lrow[r];
#pragma unroll
  for (int nt = 0; nt < 4; ++nt)
#pragma unroll
    for (int r = 0; r < 8; ++r) {
      const int srow = q0 + r + rhalf;
      const int col = hh * HDn + nt * 16 + (lane & 15);
      ctx[((size_t)bb * Sn + srow) * Dn + col] = (_Float16)(acc[nt][r] * lrow[r]);
    }
}

// ---------------------------------------------------------------------------
extern "C" void kernel_launch(void* const* d_in, const int* in_sizes, int n_in,
                              void* d_out, int out_size, void* d_ws, size_t ws_size,
                              hipStream_t stream) {
  (void)in_sizes; (void)n_in; (void)out_size; (void)ws_size;
  const float* X    = (const float*)d_in[0];
  const float* mask = (const float*)d_in[1];
  const float* Wq = (const float*)d_in[2];  const float* bq = (const float*)d_in[3];
  const float* Wk = (const float*)d_in[4];  const float* bk = (const float*)d_in[5];
  const float* Wv = (const float*)d_in[6];  const float* bv = (const float*)d_in[7];
  const float* Wo = (const float*)d_in[8];  const float* bo = (const float*)d_in[9];

  _Float16* XH   = (_Float16*)d_ws;            // 8M elem
  _Float16* WQT  = XH  + XEL;                  // 1M elem each, transposed f16
  _Float16* WKT  = WQT + WEL;
  _Float16* WVT  = WKT + WEL;
  _Float16* WOT  = WVT + WEL;
  _Float16* QH   = WOT + WEL;                  // [B,H,S,HD]
  _Float16* KH   = QH  + XEL;                  // [B,H,S,HD]
  _Float16* VTH  = KH  + XEL;                  // [B,H,HD,S]
  _Float16* CTXH = VTH + XEL;                  // [B,S,D]

  // 1) precision conversion (+ weight transpose)
  cvt_f16_kernel<<<4096, 256, 0, stream>>>(X, XH, (int)XEL);
  cvt_wt_kernel<<<(int)(WEL / 256), 256, 0, stream>>>(Wq, WQT);
  cvt_wt_kernel<<<(int)(WEL / 256), 256, 0, stream>>>(Wk, WKT);
  cvt_wt_kernel<<<(int)(WEL / 256), 256, 0, stream>>>(Wv, WVT);
  cvt_wt_kernel<<<(int)(WEL / 256), 256, 0, stream>>>(Wo, WOT);

  // 2) QKV projections (WMMA GEMMs)
  dim3 ggrid(Mn / 128, Dn / 128);
  gemm_wmma_kernel<<<ggrid, 256, 0, stream>>>(XH, WQT, bq, QH, 0);
  gemm_wmma_kernel<<<ggrid, 256, 0, stream>>>(XH, WKT, bk, KH, 0);
  gemm_wmma_kernel<<<ggrid, 256, 0, stream>>>(XH, WVT, bv, VTH, 1);

  // 3) flash attention
  dim3 agrid(Bn * Hn, Sn / 128);
  attn_wmma_kernel<<<agrid, 256, 0, stream>>>(QH, KH, VTH, mask, CTXH);

  // 4) output projection -> f32 d_out
  gemm_wmma_kernel<<<ggrid, 256, 0, stream>>>(CTXH, WOT, bo, (float*)d_out, 2);
}
